// TiPNN_37460704756394
// MI455X (gfx1250) — matrix operand order repaired and checked by reading, not compiled, not run on hardware
//
#include <hip/hip_runtime.h>
#include <hip/hip_bf16.h>
#include <hip/hip_fp16.h>

typedef __attribute__((ext_vector_type(16))) _Float16 v16h;
typedef __attribute__((ext_vector_type(8)))  float    v8f;

#define B_    4
#define NEG_  34
#define N_    20000
#define E_    150000
#define D_    128
#define R_    400
#define L_    3
#define K13   (13 * D_)   // 1664
#define F_    (2 * D_)    // 256
#define EPS_  1e-6f

// ---------- float atomic max/min via ordered-int trick ----------
__device__ inline void atomicMaxF(float* a, float v) {
    if (v >= 0.0f) atomicMax((int*)a, __float_as_int(v));
    else           atomicMin((unsigned int*)a, __float_as_uint(v));
}
__device__ inline void atomicMinF(float* a, float v) {
    if (v >= 0.0f) atomicMin((int*)a, __float_as_int(v));
    else           atomicMax((unsigned int*)a, __float_as_uint(v));
}

// ---------- query / h2 / t2 / r2 prep ----------
__global__ void k_prep(const int* __restrict__ qt, const float* __restrict__ qw,
                       int* __restrict__ t2, int* __restrict__ h2_0,
                       float* __restrict__ query) {
    int b = blockIdx.x;
    __shared__ int s_is, s_r2;
    if (threadIdx.x == 0) {
        int h0 = qt[(b * NEG_) * 3 + 0];
        int all_eq = 1;
        for (int j = 1; j < NEG_; ++j)
            all_eq &= (qt[(b * NEG_ + j) * 3 + 0] == h0);
        int r0 = qt[(b * NEG_) * 3 + 1];
        int t0 = qt[(b * NEG_) * 3 + 2];
        s_is = all_eq;
        s_r2 = all_eq ? r0 : (r0 + R_ / 2);
        h2_0[b] = all_eq ? h0 : t0;
    }
    __syncthreads();
    for (int j = threadIdx.x; j < NEG_; j += blockDim.x) {
        int hj = qt[(b * NEG_ + j) * 3 + 0];
        int tj = qt[(b * NEG_ + j) * 3 + 2];
        t2[b * NEG_ + j] = s_is ? tj : hj;
    }
    for (int d = threadIdx.x; d < D_; d += blockDim.x)
        query[b * D_ + d] = qw[(long)s_r2 * D_ + d];
}

// ---------- x <- boundary ----------
__global__ void k_init_x(float* __restrict__ x, const int* __restrict__ h2_0,
                         const float* __restrict__ query) {
    long i = blockIdx.x * (long)blockDim.x + threadIdx.x;
    if (i >= (long)B_ * N_ * D_) return;
    int d = (int)(i % D_);
    long nd = i / D_;
    int n = (int)(nd % N_);
    int b = (int)(nd / N_);
    x[i] = (n == h2_0[b]) ? query[b * D_ + d] : 0.0f;
}

// ---------- seed accumulators with the boundary "message" ----------
__global__ void k_init_accum(float* __restrict__ sum, float* __restrict__ sumsq,
                             float* __restrict__ mx, float* __restrict__ mn,
                             float* __restrict__ deg, float* __restrict__ logsum,
                             const int* __restrict__ h2_0, const float* __restrict__ query) {
    long i = blockIdx.x * (long)blockDim.x + threadIdx.x;
    if (i >= (long)B_ * N_ * D_) return;
    int d = (int)(i % D_);
    long nd = i / D_;
    int n = (int)(nd % N_);
    int b = (int)(nd / N_);
    float bv = (n == h2_0[b]) ? query[b * D_ + d] : 0.0f;
    sum[i] = bv; sumsq[i] = bv * bv; mx[i] = bv; mn[i] = bv;
    if (b == 0 && d == 0) deg[n] = 1.0f;   // boundary message counts once per node
    if (i == 0) logsum[0] = 0.0f;
}

// ---------- rel = query @ rel_W.T + rel_b   (B x R*D) ----------
__global__ void k_rel(const float* __restrict__ query, const float* __restrict__ relW,
                      const float* __restrict__ relb, float* __restrict__ rel) {
    long i = blockIdx.x * (long)blockDim.x + threadIdx.x;
    if (i >= (long)B_ * R_ * D_) return;
    int j = (int)(i % (R_ * D_));
    int b = (int)(i / (R_ * D_));
    const float* w = relW + (long)j * D_;
    const float* q = query + b * D_;
    float acc = relb[j];
#pragma unroll 8
    for (int k = 0; k < D_; ++k) acc += q[k] * w[k];
    rel[i] = acc;
}

// ---------- degree count ----------
__global__ void k_deg(const int* __restrict__ dst, float* __restrict__ deg) {
    int e = blockIdx.x * blockDim.x + threadIdx.x;
    if (e < E_) atomicAdd(&deg[dst[e]], 1.0f);
}

// ---------- per-edge message + segment scatter (sum/sumsq/max/min) ----------
__global__ void k_scatter(const float* __restrict__ x, const float* __restrict__ rel,
                          const int* __restrict__ src, const int* __restrict__ dst,
                          const int* __restrict__ etype, const int* __restrict__ etime,
                          const float* __restrict__ tw, const float* __restrict__ tb,
                          float* __restrict__ sum, float* __restrict__ sumsq,
                          float* __restrict__ mx, float* __restrict__ mn) {
    int e = blockIdx.x;
    int d = threadIdx.x;                      // 128 threads == D
    int te = etype[e];
    float tt = (float)etime[e];
    float temb = cosf(tt * tw[(long)te * D_ + d] + tb[(long)te * D_ + d]);
    int s = src[e], t = dst[e];
#pragma unroll
    for (int b = 0; b < B_; ++b) {
        float xv = x[((long)b * N_ + s) * D_ + d];
        float rv = rel[((long)b * R_ + te) * D_ + d];
        float m = xv * (rv + temb);
        long o = ((long)b * N_ + t) * D_ + d;
        atomicAdd(&sum[o], m);
        atomicAdd(&sumsq[o], m * m);
        atomicMaxF(&mx[o], m);
        atomicMinF(&mn[o], m);
    }
}

// ---------- sum of log(deg) ----------
__global__ void k_logsum(const float* __restrict__ deg, float* __restrict__ logsum) {
    int n = blockIdx.x * blockDim.x + threadIdx.x;
    if (n < N_) atomicAdd(logsum, logf(deg[n]));
}

// ---------- lin_W (f32) -> f16 ----------
__global__ void k_wh16(const float* __restrict__ w, _Float16* __restrict__ wh, int n) {
    int i = blockIdx.x * blockDim.x + threadIdx.x;
    if (i < n) wh[i] = (_Float16)w[i];
}

// ---------- fused feature build + (16 nodes x 128 out) WMMA GEMM + ReLU ----------
__global__ void __launch_bounds__(256)
k_gemm(const float* __restrict__ x, const float* __restrict__ sum,
       const float* __restrict__ sumsq, const float* __restrict__ mx,
       const float* __restrict__ mn, const float* __restrict__ deg,
       const float* __restrict__ logsum, const _Float16* __restrict__ wh,
       const float* __restrict__ linb, float* __restrict__ xout) {
    __shared__ _Float16 As[16][136];          // 16 nodes x 128-k chunk (padded)
    int b   = blockIdx.y;
    int n0  = blockIdx.x * 16;
    int tid = threadIdx.x;
    int wave = tid >> 5;                      // 8 waves -> 8 output tiles of 16
    int lane = tid & 31;
    int hl   = lane >> 4;                     // half-wave select
    int mrow = lane & 15;
    float logmean = logsum[0] / (float)N_;

    v8f c = {};
    for (int chunk = 0; chunk < K13; chunk += 128) {
        // --- stage 16x128 f16 feature tile into LDS ---
        for (int idx = tid; idx < 16 * 128; idx += 256) {
            int ni = idx >> 7;
            int kk = idx & 127;
            int gk = chunk + kk;
            int node = n0 + ni;
            float val;
            if (gk < D_) {
                val = x[((long)b * N_ + node) * D_ + gk];
            } else {
                int k2 = gk - D_;             // 0 .. 12D-1, layout f*3 + s
                int f  = k2 / 3;
                int s  = k2 - 3 * f;
                float dg  = deg[node];
                float inv = 1.0f / dg;
                int fd = f & (D_ - 1);
                long o = ((long)b * N_ + node) * D_ + fd;
                float feat;
                int fs = f >> 7;              // 0:mean 1:max 2:min 3:std
                if (fs == 0)      feat = sum[o] * inv;
                else if (fs == 1) feat = mx[o];
                else if (fs == 2) feat = mn[o];
                else {
                    float mean = sum[o] * inv;
                    feat = sqrtf(fmaxf(sumsq[o] * inv - mean * mean, EPS_));
                }
                float sc = logf(dg) / logmean;
                float sv = (s == 0) ? 1.0f : (s == 1 ? sc : 1.0f / fmaxf(sc, 0.01f));
                val = feat * sv;
            }
            As[ni][kk] = (_Float16)val;
        }
        __syncthreads();
        // --- 4 x v_wmma_f32_16x16x32_f16 over this chunk ---
#pragma unroll
        for (int ks = 0; ks < 128; ks += 32) {
            v16h a, bf;
            // A 16x32 f16, k order of this code (matches B): lanes 0-15 hold K 0-15; lanes 16-31 K 16-31
#pragma unroll
            for (int i = 0; i < 8; ++i) {
                a[i]     = As[mrow][ks + hl * 16 + i];
                a[i + 8] = As[mrow][ks + hl * 16 + 8 + i];
            }
            // B 32x16 f16: lane&15 = column (output dim); half-wave selects K 0-15 / 16-31
            int col = wave * 16 + mrow;
            const _Float16* bp = wh + (long)col * K13 + chunk + ks + hl * 16;
#pragma unroll
            for (int j = 0; j < 16; ++j) bf[j] = bp[j];
            c = __builtin_amdgcn_wmma_f32_16x16x32_f16(false, a, false, bf,
                                                       (short)0, c, false, false);
        }
        __syncthreads();
    }
    // C/D layout: VGPR r -> M = r (lanes 0-15) / r+8 (lanes 16-31); N = lane&15
#pragma unroll
    for (int r = 0; r < 8; ++r) {
        int node = n0 + r + hl * 8;
        int od   = wave * 16 + mrow;
        float v  = c[r] + linb[od];
        xout[((long)b * N_ + node) * D_ + od] = fmaxf(v, 0.0f);
    }
}

// ---------- final per-(b,neg) 2-layer MLP readout ----------
__global__ void k_readout(const float* __restrict__ x, const float* __restrict__ query,
                          const int* __restrict__ t2,
                          const float* __restrict__ W1, const float* __restrict__ b1,
                          const float* __restrict__ W2, const float* __restrict__ b2,
                          float* __restrict__ out) {
    __shared__ float ft[F_];
    __shared__ float red[256];
    int bj = blockIdx.x;
    int b = bj / NEG_, j = bj % NEG_;
    int node = t2[b * NEG_ + j];
    int i = threadIdx.x;                      // 256 threads
    ft[i] = (i < D_) ? x[((long)b * N_ + node) * D_ + i]
                     : query[b * D_ + (i - D_)];
    __syncthreads();
    float acc = b1[i];
    const float* w = W1 + (long)i * F_;
#pragma unroll 8
    for (int k = 0; k < F_; ++k) acc += ft[k] * w[k];
    red[i] = fmaxf(acc, 0.0f) * W2[i];
    __syncthreads();
    for (int sft = 128; sft > 0; sft >>= 1) {
        if (i < sft) red[i] += red[i + sft];
        __syncthreads();
    }
    if (i == 0) out[bj] = red[0] + b2[0];
}

extern "C" void kernel_launch(void* const* d_in, const int* in_sizes, int n_in,
                              void* d_out, int out_size, void* d_ws, size_t ws_size,
                              hipStream_t stream) {
    const int*   qt    = (const int*)d_in[0];
    const int*   eidx  = (const int*)d_in[1];
    const int*   etype = (const int*)d_in[2];
    const int*   etime = (const int*)d_in[3];
    const float* qw    = (const float*)d_in[4];
    const float* relW  = (const float*)d_in[5];
    const float* relb  = (const float*)d_in[6];
    const float* tw    = (const float*)d_in[7];
    const float* tb    = (const float*)d_in[8];
    const float* linW  = (const float*)d_in[9];
    const float* linb  = (const float*)d_in[10];
    const float* W1    = (const float*)d_in[11];
    const float* b1    = (const float*)d_in[12];
    const float* W2    = (const float*)d_in[13];
    const float* b2    = (const float*)d_in[14];

    char* ws = (char*)d_ws;
    size_t off = 0;
    auto alloc = [&](size_t bytes) -> void* {
        void* p = ws + off;
        off += (bytes + 255) & ~((size_t)255);
        return p;
    };
    const long BND = (long)B_ * N_ * D_;
    float*     query  = (float*)alloc((size_t)B_ * D_ * 4);
    int*       t2     = (int*)  alloc((size_t)B_ * NEG_ * 4);
    int*       h2     = (int*)  alloc((size_t)B_ * 4);
    float*     deg    = (float*)alloc((size_t)N_ * 4);
    float*     logsum = (float*)alloc(256);
    float*     rel    = (float*)alloc((size_t)B_ * R_ * D_ * 4);
    float*     x0     = (float*)alloc((size_t)BND * 4);
    float*     x1     = (float*)alloc((size_t)BND * 4);
    float*     sum    = (float*)alloc((size_t)BND * 4);
    float*     sumsq  = (float*)alloc((size_t)BND * 4);
    float*     mx     = (float*)alloc((size_t)BND * 4);
    float*     mn     = (float*)alloc((size_t)BND * 4);
    _Float16*  wh     = (_Float16*)alloc((size_t)D_ * K13 * 2);

    const int* src = eidx;
    const int* dst = eidx + E_;

    k_prep<<<B_, 128, 0, stream>>>(qt, qw, t2, h2, query);

    long tot = BND;
    int  blks = (int)((tot + 255) / 256);
    k_init_x<<<blks, 256, 0, stream>>>(x0, h2, query);

    float* xa = x0;
    float* xb = x1;
    for (int l = 0; l < L_; ++l) {
        k_rel<<<(B_ * R_ * D_ + 255) / 256, 256, 0, stream>>>(
            query, relW + (long)l * R_ * D_ * D_, relb + (long)l * R_ * D_, rel);
        k_init_accum<<<blks, 256, 0, stream>>>(sum, sumsq, mx, mn, deg, logsum, h2, query);
        k_deg<<<(E_ + 255) / 256, 256, 0, stream>>>(dst, deg);
        k_scatter<<<E_, 128, 0, stream>>>(xa, rel, src, dst, etype, etime,
                                          tw + (long)l * R_ * D_, tb + (long)l * R_ * D_,
                                          sum, sumsq, mx, mn);
        k_logsum<<<(N_ + 255) / 256, 256, 0, stream>>>(deg, logsum);
        k_wh16<<<(D_ * K13 + 255) / 256, 256, 0, stream>>>(linW + (long)l * D_ * K13, wh, D_ * K13);
        dim3 g(N_ / 16, B_);
        k_gemm<<<g, 256, 0, stream>>>(xa, sum, sumsq, mx, mn, deg, logsum, wh,
                                      linb + (long)l * D_, xb);
        float* tswap = xa; xa = xb; xb = tswap;
    }

    k_readout<<<B_ * NEG_, 256, 0, stream>>>(xa, query, t2, W1, b1, W2, b2, (float*)d_out);
}
